// MaskedCausalAttention_27771258536589
// MI455X (gfx1250) — compile-verified, matrix operand order checked by reading, not run on hardware
//
#include <hip/hip_runtime.h>

// ---------------------------------------------------------------------------
// MI455X (gfx1250) causal MHA.  All matmuls via v_wmma_f32_16x16x32_bf16.
// Compute-bound (~137 GFLOP, ~150MB L2-resident data): fp32 -> bf16 operands,
// fp32 accumulation.  All kernels software-pipeline global->LDS staging
// (double-buffered LDS, one barrier per K-step) so tile movement hides behind
// the WMMA stream.
// ---------------------------------------------------------------------------

typedef __bf16 bf16;
typedef __attribute__((ext_vector_type(16))) __bf16 v16bf;
typedef __attribute__((ext_vector_type(8)))  float  v8f;

#define H_DIM 1024
#define N_HEADS 16
#define BATCH 4
#define SEQ 2048
#define DHEAD 64
#define MTOT (BATCH * SEQ) // 8192

#define TM 128
#define TN 128
#define TK 32

union FragU { v16bf v; uint4 q[2]; };
union Pack4 { bf16 h[4]; uint2 u; };

// CDNA5 wave32 16-bit A/B fragment gather from row-major [rows][ld] LDS.
// Lane l holds row (l&15); element e -> col c0 + ((e&8)<<1) + (l>>4)*8 + (e&7)
// => two contiguous 16-byte chunks (ds_load_b128 x2).
__device__ __forceinline__ v16bf load_frag(const bf16* base, int row, int ld,
                                           int c0, int hi) {
  const bf16* p = base + (size_t)row * ld + c0 + hi * 8;
  FragU f;
  f.q[0] = *(const uint4*)(p);
  f.q[1] = *(const uint4*)(p + 16);
  return f.v;
}

__device__ __forceinline__ v8f wmma_bf16(v16bf a, v16bf b, v8f c) {
  return __builtin_amdgcn_wmma_f32_16x16x32_bf16(false, a, false, b,
                                                 (short)0, c, false, false);
}

__device__ __forceinline__ float half_max16(float v) {
  v = fmaxf(v, __shfl_xor(v, 1, 32));
  v = fmaxf(v, __shfl_xor(v, 2, 32));
  v = fmaxf(v, __shfl_xor(v, 4, 32));
  v = fmaxf(v, __shfl_xor(v, 8, 32));
  return v;
}
__device__ __forceinline__ float half_sum16(float v) {
  v += __shfl_xor(v, 1, 32);
  v += __shfl_xor(v, 2, 32);
  v += __shfl_xor(v, 4, 32);
  v += __shfl_xor(v, 8, 32);
  return v;
}

// ---------------------------------------------------------------------------
// Kernel 1: fused QKV projection.  Y = (x @ W + b) [* 1/8 for Q].
// blockIdx.z selects Q/K/V.  Block tile 128x128, 8 waves as 4(M)x2(N), each
// wave 32x64 = 8 accumulators.  Double-buffered LDS, 1 barrier per K-step.
// Q,K stored [b,h,t,d]; V stored transposed [b,h,d,t] for the PV operand.
// ---------------------------------------------------------------------------
__global__ __launch_bounds__(256) void qkv_gemm(
    const float* __restrict__ x,
    const float* __restrict__ Wq, const float* __restrict__ Wk,
    const float* __restrict__ Wv,
    const float* __restrict__ bq, const float* __restrict__ bk,
    const float* __restrict__ bv,
    bf16* __restrict__ q_ws, bf16* __restrict__ k_ws, bf16* __restrict__ v_ws) {
  __shared__ alignas(16) bf16 sA[2][TM * TK];   // x tile  [m][k]
  __shared__ alignas(16) bf16 sB[2][TN * TK];   // W tile transposed [n][k]

  const int z = blockIdx.z;
  const float* W    = (z == 0) ? Wq : (z == 1) ? Wk : Wv;
  const float* bias = (z == 0) ? bq : (z == 1) ? bk : bv;

  const int m0 = blockIdx.x * TM;
  const int n0 = blockIdx.y * TN;
  const int tid = threadIdx.x;
  const int wave = tid >> 5, lane = tid & 31;
  const int wm = wave >> 1, wn = wave & 1;
  const int hi = lane >> 4, ln = lane & 15;

  float4 ra[4];      // A staging: 4 x float4 per thread (128x32 fp32)
  float  rb[4][4];   // B staging: 4 x (n, 4 consecutive k) per thread

  auto fetch = [&](int k0) {
#pragma unroll
    for (int q = 0; q < 4; ++q) {
      int t = q * 256 + tid;
      int rowA = t >> 3, c4 = t & 7;                 // 8 float4 per A row
      ra[q] = *(const float4*)(x + (size_t)(m0 + rowA) * H_DIM + k0 + c4 * 4);
      int nB = t & 127, kq = t >> 7;                 // coalesced along n
#pragma unroll
      for (int u = 0; u < 4; ++u)
        rb[q][u] = W[(size_t)(k0 + kq * 4 + u) * H_DIM + n0 + nB];
    }
  };
  auto stage = [&](int buf) {
#pragma unroll
    for (int q = 0; q < 4; ++q) {
      int t = q * 256 + tid;
      Pack4 pa;
      pa.h[0] = (bf16)ra[q].x; pa.h[1] = (bf16)ra[q].y;
      pa.h[2] = (bf16)ra[q].z; pa.h[3] = (bf16)ra[q].w;
      *(uint2*)(&sA[buf][t * 4]) = pa.u;             // linear [m][k]
      int nB = t & 127, kq = t >> 7;
      Pack4 pb;
      pb.h[0] = (bf16)rb[q][0]; pb.h[1] = (bf16)rb[q][1];
      pb.h[2] = (bf16)rb[q][2]; pb.h[3] = (bf16)rb[q][3];
      *(uint2*)(&sB[buf][nB * TK + kq * 4]) = pb.u;  // transposed [n][k]
    }
  };

  v8f acc[2][4] = {};

  fetch(0);
  stage(0);
  __syncthreads();
  int cur = 0;

  for (int k0 = 0; k0 < H_DIM; k0 += TK) {
    const bool nxt = (k0 + TK) < H_DIM;
    if (nxt) fetch(k0 + TK);   // loads in flight during WMMAs below

    const bf16* A = sA[cur];
    const bf16* B = sB[cur];
    v16bf af0 = load_frag(A, wm * 32 + ln, TK, 0, hi);
    v16bf af1 = load_frag(A, wm * 32 + 16 + ln, TK, 0, hi);
#pragma unroll
    for (int s = 0; s < 4; ++s) {
      v16bf bfm = load_frag(B, wn * 64 + s * 16 + ln, TK, 0, hi);
      acc[0][s] = wmma_bf16(af0, bfm, acc[0][s]);
      acc[1][s] = wmma_bf16(af1, bfm, acc[1][s]);
    }

    if (nxt) {
      stage(cur ^ 1);          // other buffer: safe while peers compute 'cur'
      __syncthreads();
      cur ^= 1;
    }
  }

  // epilogue: bias (+ fold 1/sqrt(64) into Q) + scatter to head-major layouts
  const float scaleq = (z == 0) ? 0.125f : 1.0f;
#pragma unroll
  for (int s = 0; s < 4; ++s) {
    int n_g = n0 + wn * 64 + s * 16 + ln;
    int head = n_g >> 6, d = n_g & 63;
    float bval = bias[n_g];
#pragma unroll
    for (int mi = 0; mi < 2; ++mi)
#pragma unroll
      for (int j = 0; j < 8; ++j) {
        int m_g = m0 + wm * 32 + mi * 16 + j + 8 * hi;
        int bb = m_g >> 11, tt = m_g & (SEQ - 1);
        float val = (acc[mi][s][j] + bval) * scaleq;
        size_t bh = (size_t)(bb * N_HEADS + head);
        if (z == 2) {
          v_ws[(bh * DHEAD + d) * SEQ + tt] = (bf16)val;  // V^T: [b,h,d,t]
        } else {
          bf16* dst = (z == 0) ? q_ws : k_ws;
          dst[(bh * SEQ + tt) * DHEAD + d] = (bf16)val;   // [b,h,t,d]
        }
      }
  }
}

// ---------------------------------------------------------------------------
// Kernel 2: causal flash attention.  Block = (b, head, 128 query rows);
// 8 waves each own 16 rows with private online softmax (row state per-j VGPR
// replicated across each 16-lane half, reduced with shfl_xor).  K/V key tiles
// of 64 double-buffered in LDS, prefetched during compute.
// ---------------------------------------------------------------------------
__global__ __launch_bounds__(256) void attn_flash(
    const bf16* __restrict__ q_ws, const bf16* __restrict__ k_ws,
    const bf16* __restrict__ v_ws, bf16* __restrict__ a_ws) {
  __shared__ alignas(16) bf16 sK[2][64 * 64];     // [s][d]
  __shared__ alignas(16) bf16 sV[2][64 * 64];     // [d][s] (from V^T)
  __shared__ alignas(16) bf16 sP[8 * 16 * 64];    // per-wave P tile [m][s]

  const int tid = threadIdx.x;
  const int wave = tid >> 5, lane = tid & 31;
  const int hi = lane >> 4, ln = lane & 15;

  const int bid = blockIdx.x;
  const int qt = bid & 15;
  const int head = (bid >> 4) & 15;
  const int b = bid >> 8;
  const int q0 = qt * 128;
  const int qw = q0 + wave * 16;

  const size_t bh = (size_t)(b * N_HEADS + head);
  const bf16* Q = q_ws + bh * SEQ * DHEAD;
  const bf16* K = k_ws + bh * SEQ * DHEAD;
  const bf16* V = v_ws + bh * DHEAD * SEQ;   // [d][t]

  uint4 rk[2], rv[2];
  auto fetchKV = [&](int s0) {
    const uint4* Kg = (const uint4*)(K + (size_t)s0 * DHEAD);
    rk[0] = Kg[tid];
    rk[1] = Kg[tid + 256];
    const bf16* Vg = V + s0;
#pragma unroll
    for (int q = 0; q < 2; ++q) {
      int i = q * 256 + tid;
      int d = i >> 3, c = (i & 7) * 8;
      rv[q] = *(const uint4*)(Vg + (size_t)d * SEQ + c);
    }
  };
  auto stageKV = [&](int buf) {
    ((uint4*)sK[buf])[tid] = rk[0];
    ((uint4*)sK[buf])[tid + 256] = rk[1];
#pragma unroll
    for (int q = 0; q < 2; ++q) ((uint4*)sV[buf])[q * 256 + tid] = rv[q];
  };

  // Q fragments (A-layout), reused across all key tiles (scale pre-folded).
  v16bf qf[2];
#pragma unroll
  for (int ks = 0; ks < 2; ++ks)
    qf[ks] = load_frag(Q, qw + ln, DHEAD, ks * 32, hi);

  float mrow[8], lrow[8];
#pragma unroll
  for (int j = 0; j < 8; ++j) { mrow[j] = -3.0e38f; lrow[j] = 0.0f; }
  v8f oacc[4] = {};

  const int nst = (q0 + 128) >> 6;
  fetchKV(0);
  stageKV(0);
  __syncthreads();
  int cur = 0;

  for (int st = 0; st < nst; ++st) {
    const int s0 = st * 64;
    const bool nxt = (st + 1) < nst;
    if (nxt) fetchKV(s0 + 64);   // prefetch next tile during compute

    if (s0 <= qw + 15) {         // tile not fully above diagonal for this wave
      const bf16* Kt = sK[cur];
      const bf16* Vt = sV[cur];
      // ---- S = Q K^T
      v8f sacc[4] = {};
#pragma unroll
      for (int s = 0; s < 4; ++s)
#pragma unroll
        for (int ks = 0; ks < 2; ++ks) {
          v16bf kf = load_frag(Kt, s * 16 + ln, DHEAD, ks * 32, hi);
          sacc[s] = wmma_bf16(qf[ks], kf, sacc[s]);
        }
      // ---- causal mask (scale already folded into Q)
#pragma unroll
      for (int s = 0; s < 4; ++s) {
        int sg = s0 + s * 16 + ln;
#pragma unroll
        for (int j = 0; j < 8; ++j) {
          int qm = qw + j + 8 * hi;
          sacc[s][j] = (sg > qm) ? -3.0e38f : sacc[s][j];
        }
      }
      // ---- online softmax update
      float corr[8];
#pragma unroll
      for (int j = 0; j < 8; ++j) {
        float v = fmaxf(fmaxf(sacc[0][j], sacc[1][j]),
                        fmaxf(sacc[2][j], sacc[3][j]));
        v = half_max16(v);
        float mn = fmaxf(mrow[j], v);
        corr[j] = __expf(mrow[j] - mn);
        mrow[j] = mn;
      }
      float rsum[8];
#pragma unroll
      for (int j = 0; j < 8; ++j) rsum[j] = 0.0f;
#pragma unroll
      for (int s = 0; s < 4; ++s)
#pragma unroll
        for (int j = 0; j < 8; ++j) {
          float p = __expf(sacc[s][j] - mrow[j]);
          sacc[s][j] = p;
          rsum[j] += p;
        }
#pragma unroll
      for (int j = 0; j < 8; ++j)
        lrow[j] = lrow[j] * corr[j] + half_sum16(rsum[j]);
#pragma unroll
      for (int t = 0; t < 4; ++t)
#pragma unroll
        for (int j = 0; j < 8; ++j) oacc[t][j] *= corr[j];

      // ---- P: C-layout -> wave-private LDS -> A-layout
      bf16* Pw = sP + wave * (16 * 64);
#pragma unroll
      for (int s = 0; s < 4; ++s)
#pragma unroll
        for (int j = 0; j < 8; ++j)
          Pw[(j + 8 * hi) * 64 + s * 16 + ln] = (bf16)sacc[s][j];
      asm volatile("s_wait_dscnt 0" ::: "memory");  // wave-local LDS RAW

      // ---- O += P V
#pragma unroll
      for (int ks = 0; ks < 2; ++ks) {
        v16bf pf = load_frag(Pw, ln, 64, ks * 32, hi);
#pragma unroll
        for (int t = 0; t < 4; ++t) {
          v16bf vf = load_frag(Vt, t * 16 + ln, 64, ks * 32, hi);
          oacc[t] = wmma_bf16(pf, vf, oacc[t]);
        }
      }
    }

    if (nxt) stageKV(cur ^ 1);   // other buffer: safe while peers compute 'cur'
    __syncthreads();
    cur ^= 1;
  }

  // ---- normalize + store attn (bf16) in [b,t, head*64+d] layout
#pragma unroll
  for (int t = 0; t < 4; ++t) {
    int d = t * 16 + ln;
#pragma unroll
    for (int j = 0; j < 8; ++j) {
      int m = qw + j + 8 * hi;
      float val = oacc[t][j] / lrow[j];
      a_ws[((size_t)(b * SEQ + m)) * H_DIM + head * DHEAD + d] = (bf16)val;
    }
  }
}

// ---------------------------------------------------------------------------
// Kernel 3: output projection.  out = attn(bf16) @ Wo + bo  (fp32 out)
// Same pipelined 128x128 tiling as kernel 1; A is already bf16.
// ---------------------------------------------------------------------------
__global__ __launch_bounds__(256) void out_gemm(
    const bf16* __restrict__ attn, const float* __restrict__ Wo,
    const float* __restrict__ bo, float* __restrict__ out) {
  __shared__ alignas(16) bf16 sA[2][TM * TK];
  __shared__ alignas(16) bf16 sB[2][TN * TK];

  const int m0 = blockIdx.x * TM;
  const int n0 = blockIdx.y * TN;
  const int tid = threadIdx.x;
  const int wave = tid >> 5, lane = tid & 31;
  const int wm = wave >> 1, wn = wave & 1;
  const int hi = lane >> 4, ln = lane & 15;

  uint4 ra[2];       // A staging: 2 x uint4 (8 bf16) per thread
  float rb[4][4];

  auto fetch = [&](int k0) {
#pragma unroll
    for (int q = 0; q < 2; ++q) {
      int i = q * 256 + tid;
      int rowA = i >> 2, c8 = (i & 3) * 8;           // 4 uint4 per A row
      ra[q] = *(const uint4*)(attn + (size_t)(m0 + rowA) * H_DIM + k0 + c8);
    }
#pragma unroll
    for (int q = 0; q < 4; ++q) {
      int t = q * 256 + tid;
      int nB = t & 127, kq = t >> 7;
#pragma unroll
      for (int u = 0; u < 4; ++u)
        rb[q][u] = Wo[(size_t)(k0 + kq * 4 + u) * H_DIM + n0 + nB];
    }
  };
  auto stage = [&](int buf) {
#pragma unroll
    for (int q = 0; q < 2; ++q)
      ((uint4*)sA[buf])[q * 256 + tid] = ra[q];
#pragma unroll
    for (int q = 0; q < 4; ++q) {
      int t = q * 256 + tid;
      int nB = t & 127, kq = t >> 7;
      Pack4 pb;
      pb.h[0] = (bf16)rb[q][0]; pb.h[1] = (bf16)rb[q][1];
      pb.h[2] = (bf16)rb[q][2]; pb.h[3] = (bf16)rb[q][3];
      *(uint2*)(&sB[buf][nB * TK + kq * 4]) = pb.u;
    }
  };

  v8f acc[2][4] = {};

  fetch(0);
  stage(0);
  __syncthreads();
  int cur = 0;

  for (int k0 = 0; k0 < H_DIM; k0 += TK) {
    const bool nxt = (k0 + TK) < H_DIM;
    if (nxt) fetch(k0 + TK);

    const bf16* A = sA[cur];
    const bf16* B = sB[cur];
    v16bf af0 = load_frag(A, wm * 32 + ln, TK, 0, hi);
    v16bf af1 = load_frag(A, wm * 32 + 16 + ln, TK, 0, hi);
#pragma unroll
    for (int s = 0; s < 4; ++s) {
      v16bf bfm = load_frag(B, wn * 64 + s * 16 + ln, TK, 0, hi);
      acc[0][s] = wmma_bf16(af0, bfm, acc[0][s]);
      acc[1][s] = wmma_bf16(af1, bfm, acc[1][s]);
    }

    if (nxt) {
      stage(cur ^ 1);
      __syncthreads();
      cur ^= 1;
    }
  }

#pragma unroll
  for (int s = 0; s < 4; ++s) {
    int n_g = n0 + wn * 64 + s * 16 + ln;
    float bval = bo[n_g];
#pragma unroll
    for (int mi = 0; mi < 2; ++mi)
#pragma unroll
      for (int j = 0; j < 8; ++j) {
        int m_g = m0 + wm * 32 + mi * 16 + j + 8 * hi;
        out[(size_t)m_g * H_DIM + n_g] = acc[mi][s][j] + bval;
      }
  }
}

// ---------------------------------------------------------------------------
extern "C" void kernel_launch(void* const* d_in, const int* in_sizes, int n_in,
                              void* d_out, int out_size, void* d_ws,
                              size_t ws_size, hipStream_t stream) {
  const float* x  = (const float*)d_in[0];
  const float* Wq = (const float*)d_in[1];
  const float* bq = (const float*)d_in[2];
  const float* Wk = (const float*)d_in[3];
  const float* bk = (const float*)d_in[4];
  const float* Wv = (const float*)d_in[5];
  const float* bv = (const float*)d_in[6];
  const float* Wo = (const float*)d_in[7];
  const float* bo = (const float*)d_in[8];
  float* out = (float*)d_out;

  const size_t QKV_ELEMS = (size_t)BATCH * N_HEADS * SEQ * DHEAD;
  bf16* q_ws = (bf16*)d_ws;
  bf16* k_ws = q_ws + QKV_ELEMS;
  bf16* v_ws = k_ws + QKV_ELEMS;
  bf16* a_ws = v_ws + QKV_ELEMS;   // 64 MB workspace total

  dim3 gQKV(MTOT / TM, H_DIM / TN, 3);
  qkv_gemm<<<gQKV, 256, 0, stream>>>(x, Wq, Wk, Wv, bq, bk, bv,
                                     q_ws, k_ws, v_ws);

  dim3 gATT(BATCH * N_HEADS * (SEQ / 128));
  attn_flash<<<gATT, 256, 0, stream>>>(q_ws, k_ws, v_ws, a_ws);

  dim3 gOUT(MTOT / TM, H_DIM / TN);
  out_gemm<<<gOUT, 256, 0, stream>>>(a_ws, Wo, bo, out);
}
